// VectorQuantizer_5480378269811
// MI455X (gfx1250) — matrix operand order repaired and checked by reading, not compiled, not run on hardware
//
#include <hip/hip_runtime.h>
#include <hip/hip_bf16.h>

#define K_EMB   1024
#define D_DIM   64
#define N_ROWS  32768
#define HW      1024        // H*W
#define BSTRIDE 65536       // D*H*W
#define STAGE_K 64          // embeddings per LDS stage (double-buffered)
#define NSTAGES (K_EMB / STAGE_K)
#define LDS_STRIDE 72       // bf16 elems per LDS codebook row (64 + 8 pad -> 144B, bank-conflict-free)

typedef __bf16 v16bf __attribute__((ext_vector_type(16)));
typedef float  v8f   __attribute__((ext_vector_type(8)));

// ---- CDNA5 async global->LDS path (ASYNCcnt), with graceful fallback -------
#if defined(__has_builtin)
#  if __has_builtin(__builtin_amdgcn_global_load_async_to_lds_b128)
#    define HAVE_ASYNC 1
#  endif
#endif
#ifndef HAVE_ASYNC
#  define HAVE_ASYNC 0
#endif

#if HAVE_ASYNC
typedef int v4i __attribute__((ext_vector_type(4)));
typedef __attribute__((address_space(1))) v4i* g_b128;   // global int4*  (per clang diag: "int __vector(4) __device__ *")
typedef __attribute__((address_space(3))) v4i* l_b128;   // LDS int4*
#  define ASYNC_B128(gp, lp) \
      __builtin_amdgcn_global_load_async_to_lds_b128((g_b128)(void*)(gp), (l_b128)(void*)(lp), 0, 0)
#  if __has_builtin(__builtin_amdgcn_s_wait_asynccnt)
#    define WAIT_ASYNC(n) __builtin_amdgcn_s_wait_asynccnt(n)
#  else
#    define WAIT_ASYNC(n) asm volatile("s_wait_asynccnt " #n ::: "memory")
#  endif
#else
#  define WAIT_ASYNC(n) ((void)0)
#endif

static __device__ __forceinline__ unsigned short f32_to_bf16_rne(float f) {
    unsigned u = __builtin_bit_cast(unsigned, f);
    unsigned r = u + 0x7fffu + ((u >> 16) & 1u);
    return (unsigned short)(r >> 16);
}
static __device__ __forceinline__ float bf16_to_f32(unsigned short h) {
    unsigned u = ((unsigned)h) << 16;
    return __builtin_bit_cast(float, u);
}

// ---------------------------------------------------------------------------
// Kernel 0: split codebook into bf16 hi/lo and compute ||e_k||^2
// ---------------------------------------------------------------------------
__global__ void vq_prep(const float* __restrict__ emb,
                        unsigned short* __restrict__ hi,
                        unsigned short* __restrict__ lo,
                        float* __restrict__ norms) {
    const int k = blockIdx.x;
    const int d = threadIdx.x;
    float e = emb[k * D_DIM + d];
    unsigned short h = f32_to_bf16_rne(e);
    unsigned short l = f32_to_bf16_rne(e - bf16_to_f32(h));
    hi[k * D_DIM + d] = h;
    lo[k * D_DIM + d] = l;
    float s = e * e;
    #pragma unroll
    for (int off = 16; off > 0; off >>= 1) s += __shfl_xor(s, off, 32);
    __shared__ float red[2];
    if ((threadIdx.x & 31) == 0) red[threadIdx.x >> 5] = s;
    __syncthreads();
    if (threadIdx.x == 0) norms[k] = red[0] + red[1];
}

// ---------------------------------------------------------------------------
// Kernel 1: WMMA distance matmul + argmin, async double-buffered LDS stages.
// ---------------------------------------------------------------------------
__global__ void __launch_bounds__(256) vq_argmin(
    const float* __restrict__ x,
    const unsigned short* __restrict__ ehi,
    const unsigned short* __restrict__ elo,
    const float* __restrict__ norms,
    int* __restrict__ idx_out)
{
    __shared__ unsigned short sh_hi[2][STAGE_K * LDS_STRIDE];
    __shared__ unsigned short sh_lo[2][STAGE_K * LDS_STRIDE];
    __shared__ float sh_norm[2][STAGE_K];

    const int tid     = threadIdx.x;
    const int lane    = tid & 31;
    const int wave    = tid >> 5;
    const int rowBase = blockIdx.x * 128 + wave * 16;

    // ---- A operand: 16 rows of x, ISA 16-bit A layout:
    //      m = lane%16,  k(e, ha) = e + 8*ha + (e>=8 ? 8 : 0),  ha = lane/16
    const int ha = lane >> 4;
    const int n  = rowBase + (lane & 15);
    const float* xrow = x + (n >> 10) * BSTRIDE + (n & 1023);

    v16bf a_hi[2], a_lo[2];
    #pragma unroll
    for (int c = 0; c < 2; ++c) {
        #pragma unroll
        for (int e = 0; e < 16; ++e) {
            const int d = c * 32 + e + 8 * ha + ((e >= 8) ? 8 : 0);
            float xv = xrow[d * HW];
            unsigned short h = f32_to_bf16_rne(xv);
            a_hi[c][e] = __builtin_bit_cast(__bf16, h);
            a_lo[c][e] = __builtin_bit_cast(__bf16, f32_to_bf16_rne(xv - bf16_to_f32(h)));
        }
    }

    // stage issuer: 64 rows * 128B per array; 16B chunks; 256 threads -> 2 passes,
    // 4 async b128 issues per thread per stage (hi + lo).
    auto stage_issue = [&](int ss, int bb) {
        #pragma unroll
        for (int p = 0; p < 2; ++p) {
            const int c = tid + p * 256;
            const int r = c >> 3, q = c & 7;
            const unsigned short* gh = ehi + (ss * STAGE_K + r) * D_DIM + q * 8;
            const unsigned short* gl = elo + (ss * STAGE_K + r) * D_DIM + q * 8;
            unsigned short* dh = &sh_hi[bb][r * LDS_STRIDE + q * 8];
            unsigned short* dl = &sh_lo[bb][r * LDS_STRIDE + q * 8];
#if HAVE_ASYNC
            ASYNC_B128(gh, dh);
            ASYNC_B128(gl, dl);
#else
            *(uint4*)dh = *(const uint4*)gh;
            *(uint4*)dl = *(const uint4*)gl;
#endif
        }
        if (tid < STAGE_K) sh_norm[bb][tid] = norms[ss * STAGE_K + tid];
    };

    float best[8];
    int   bidx[8];
    #pragma unroll
    for (int r = 0; r < 8; ++r) { best[r] = 3.4e38f; bidx[r] = 0; }

    stage_issue(0, 0);                                    // prologue

    for (int s = 0; s < NSTAGES; ++s) {
        const int buf = s & 1;
        if (s + 1 < NSTAGES) {
            stage_issue(s + 1, buf ^ 1);                  // overlap next stage
            WAIT_ASYNC(4);                                // this wave's stage-s copies done
        } else {
            WAIT_ASYNC(0);
        }
        __syncthreads();                                  // stage s visible to all waves

        #pragma unroll 1
        for (int t = 0; t < STAGE_K / 16; ++t) {
            // B operand, ISA 16-bit B layout: n = lane%16, k(e, hb) = e + 8*hb + (e>=8 ? 8 : 0)
            const int ne = t * 16 + (lane & 15);
            const int hb = lane >> 4;
            union V { v16bf v; uint4 q[2]; } bh0, bh1, bl0, bl1;
            const uint4* ph = (const uint4*)(&sh_hi[buf][ne * LDS_STRIDE]);
            const uint4* pl = (const uint4*)(&sh_lo[buf][ne * LDS_STRIDE]);
            bh0.q[0] = ph[hb]; bh0.q[1] = ph[2 + hb];         // d in [ 0,32)
            bh1.q[0] = ph[4 + hb]; bh1.q[1] = ph[6 + hb]; // d in [32,64)
            bl0.q[0] = pl[hb]; bl0.q[1] = pl[2 + hb];
            bl1.q[0] = pl[4 + hb]; bl1.q[1] = pl[6 + hb];

            v8f acc = {0.f, 0.f, 0.f, 0.f, 0.f, 0.f, 0.f, 0.f};
            acc = __builtin_amdgcn_wmma_f32_16x16x32_bf16(false, a_hi[0], false, bh0.v, (short)0, acc, false, false);
            acc = __builtin_amdgcn_wmma_f32_16x16x32_bf16(false, a_hi[1], false, bh1.v, (short)0, acc, false, false);
            acc = __builtin_amdgcn_wmma_f32_16x16x32_bf16(false, a_hi[0], false, bl0.v, (short)0, acc, false, false);
            acc = __builtin_amdgcn_wmma_f32_16x16x32_bf16(false, a_hi[1], false, bl1.v, (short)0, acc, false, false);
            acc = __builtin_amdgcn_wmma_f32_16x16x32_bf16(false, a_lo[0], false, bh0.v, (short)0, acc, false, false);
            acc = __builtin_amdgcn_wmma_f32_16x16x32_bf16(false, a_lo[1], false, bh1.v, (short)0, acc, false, false);

            const float nrm  = sh_norm[buf][ne];
            const int   gidx = s * STAGE_K + ne;
            #pragma unroll
            for (int r = 0; r < 8; ++r) {
                float dist = nrm - 2.0f * acc[r];   // ||x||^2 is row-constant, dropped
                if (dist < best[r]) { best[r] = dist; bidx[r] = gidx; }
            }
        }
        __syncthreads();                                  // buf free for stage s+2 issue
    }

    // reduce across the 16 lanes that share each row group (wave32 butterfly)
    #pragma unroll
    for (int r = 0; r < 8; ++r) {
        #pragma unroll
        for (int off = 1; off < 16; off <<= 1) {
            float ov = __shfl_xor(best[r], off, 32);
            int   oi = __shfl_xor(bidx[r], off, 32);
            if (ov < best[r] || (ov == best[r] && oi < bidx[r])) { best[r] = ov; bidx[r] = oi; }
        }
    }
    if ((lane & 15) == 0) {
        #pragma unroll
        for (int r = 0; r < 8; ++r)
            idx_out[rowBase + r + 8 * (lane >> 4)] = bidx[r];
    }
}

// ---------------------------------------------------------------------------
// Kernel 2: gather quantized rows -> transposed out, loss sum, counts, one-hot
// ---------------------------------------------------------------------------
__global__ void __launch_bounds__(256) vq_gather(
    const float* __restrict__ x,
    const float* __restrict__ emb,
    const int* __restrict__ idx,
    float* __restrict__ out,
    float* __restrict__ encoded,
    int* __restrict__ counts,
    float* __restrict__ loss_sum)
{
    const int n = blockIdx.x * blockDim.x + threadIdx.x;
    const int k = idx[n];
    const float* er = emb + k * D_DIM;
    const float* xr = x   + (n >> 10) * BSTRIDE + (n & 1023);
    float*     orow = out + (n >> 10) * BSTRIDE + (n & 1023);
    float s = 0.f;
    #pragma unroll
    for (int d = 0; d < D_DIM; ++d) {
        float q  = er[d];
        float dv = q - xr[d * HW];
        s += dv * dv;
        orow[d * HW] = q;
    }
    atomicAdd(&counts[k], 1);
    encoded[(size_t)n * K_EMB + k] = 1.0f;

    #pragma unroll
    for (int off = 16; off > 0; off >>= 1) s += __shfl_xor(s, off, 32);
    __shared__ float red[8];
    if ((threadIdx.x & 31) == 0) red[threadIdx.x >> 5] = s;
    __syncthreads();
    if (threadIdx.x < 8) {
        float v = red[threadIdx.x];
        #pragma unroll
        for (int off = 4; off > 0; off >>= 1) v += __shfl_xor(v, off, 32);
        if (threadIdx.x == 0) atomicAdd(loss_sum, v);
    }
}

// ---------------------------------------------------------------------------
// Kernel 3: scalars (vq_loss, perplexity)
// ---------------------------------------------------------------------------
__global__ void vq_finalize(const int* __restrict__ counts,
                            const float* __restrict__ loss_sum,
                            float* __restrict__ scalars) {
    const int k = threadIdx.x;                       // 1024 threads
    float p = (float)counts[k] * (1.0f / 32768.0f);
    float t = -p * logf(p + 1e-10f);
    #pragma unroll
    for (int off = 16; off > 0; off >>= 1) t += __shfl_xor(t, off, 32);
    __shared__ float red[32];
    if ((k & 31) == 0) red[k >> 5] = t;
    __syncthreads();
    if (k < 32) {
        float v = red[k];
        #pragma unroll
        for (int off = 16; off > 0; off >>= 1) v += __shfl_xor(v, off, 32);
        if (k == 0) {
            scalars[0] = loss_sum[0] * (1.25f / 2097152.0f);  // (1+BETA)*mean
            scalars[1] = expf(v);
        }
    }
}

extern "C" void kernel_launch(void* const* d_in, const int* in_sizes, int n_in,
                              void* d_out, int out_size, void* d_ws, size_t ws_size,
                              hipStream_t stream) {
    (void)in_sizes; (void)n_in; (void)out_size; (void)ws_size;
    const float* x   = (const float*)d_in[0];
    const float* emb = (const float*)d_in[1];

    float* out     = (float*)d_out;                  // [B,D,H,W]
    float* scalars = out + 2097152;                  // vq_loss, perplexity
    float* encoded = out + 2097154;                  // [N,K] one-hot

    char* ws = (char*)d_ws;
    unsigned short* hi    = (unsigned short*)(ws + 0);       // K*D bf16
    unsigned short* lo    = (unsigned short*)(ws + 131072);  // K*D bf16
    float*          norms = (float*)(ws + 262144);           // K f32
    int*            idx   = (int*)(ws + 266240);             // N i32
    int*            counts= (int*)(ws + 397312);             // K i32
    float*          lsum  = (float*)(ws + 401408);           // 1 f32

    (void)hipMemsetAsync(counts, 0, K_EMB * sizeof(int) + sizeof(float), stream); // counts + lsum
    (void)hipMemsetAsync(encoded, 0, (size_t)N_ROWS * K_EMB * sizeof(float), stream);

    vq_prep    <<<K_EMB,        64,  0, stream>>>(emb, hi, lo, norms);
    vq_argmin  <<<N_ROWS / 128, 256, 0, stream>>>(x, hi, lo, norms, idx);
    vq_gather  <<<N_ROWS / 256, 256, 0, stream>>>(x, emb, idx, out, encoded, counts, lsum);
    vq_finalize<<<1,            1024,0, stream>>>(counts, lsum, scalars);
}
